// Expert_1743756722200
// MI455X (gfx1250) — matrix operand order repaired and checked by reading, not compiled, not run on hardware
//
#include <hip/hip_runtime.h>

// ---------------------------------------------------------------------------
// MoE expert FFN for MI455X (gfx1250, wave32, WMMA bf16 16x16x32, f32 accum)
//   pass1: h = relu(x @ w1^T + b1)   [E,1024,4096]  stored bf16 in d_ws
//   pass2: y = h @ w2^T + b2         [E,1024,1024]  f32 to d_out
// Ping-pong LDS + ping-pong staging regs; pass-2 A tiles (already bf16) are
// copied global->LDS with GLOBAL_LOAD_ASYNC_TO_LDS_B128 when available.
// ---------------------------------------------------------------------------

#if __has_builtin(__builtin_amdgcn_global_load_async_to_lds_b128)
#define HAVE_ASYNC_LDS 1
#else
#define HAVE_ASYNC_LDS 0
#endif

typedef __attribute__((ext_vector_type(16))) __bf16 v16bf;
typedef __attribute__((ext_vector_type(8)))  float  v8f;
typedef int v4i __attribute__((vector_size(16)));   // builtin's pointee type

union FragU {
    uint4  u[2];
    v16bf  v;
};

__device__ __forceinline__ void wait_async0() {
#if __has_builtin(__builtin_amdgcn_s_wait_asynccnt)
    __builtin_amdgcn_s_wait_asynccnt(0);
#else
    asm volatile("s_wait_asynccnt 0x0" ::: "memory");
#endif
}

__device__ __forceinline__ unsigned pk_bf16x2(float lo, float hi) {
#if __has_builtin(__builtin_amdgcn_perm)
    // round-to-nearest (ties up), then grab the two high halves in one
    // v_perm_b32: result = { hi[31:16], lo[31:16] }
    unsigned a = __float_as_uint(lo) + 0x8000u;
    unsigned b = __float_as_uint(hi) + 0x8000u;
    return __builtin_amdgcn_perm(b, a, 0x07060302u);
#else
    unsigned a = __float_as_uint(lo);
    unsigned b = __float_as_uint(hi);
    a = (a + 0x7FFFu + ((a >> 16) & 1u)) >> 16;
    b = (b + 0x7FFFu + ((b >> 16) & 1u)) >> 16;
    return (a & 0xFFFFu) | (b << 16);
#endif
}

__device__ __forceinline__ unsigned short f2bf(float f) {
    return (unsigned short)((__float_as_uint(f) + 0x8000u) >> 16);
}

__device__ __forceinline__ uint4 cvt8(const float4& x, const float4& y) {
    return make_uint4(pk_bf16x2(x.x, x.y), pk_bf16x2(x.z, x.w),
                      pk_bf16x2(y.x, y.y), pk_bf16x2(y.z, y.w));
}

constexpr int BM = 128;        // token tile
constexpr int BN = 128;        // output-feature tile
constexpr int BK = 32;         // K step = WMMA bf16 K
constexpr int LDSS = 40;       // LDS row stride in halfwords (32 + 8 pad, 80B)

struct StageRegs {             // unused fields are SROA'd away per template
    float4 aF[4];
    uint4  aH[2];
    float4 bF[4];
};

// Fragment loads follow the CDNA5 16-bit A/B VGPR layouts (ISA 7.12.2):
//  A 16x32: lanes 0-15 hold K{0..7,16..23}, lanes 16-31 hold K{8..15,24..31}
//  B 32x16: lane = N (mod 16), same K split as A (lanes 0-15 K{0..7,16..23}, 16-31 K{8..15,24..31})
__device__ __forceinline__ void compute_stage(const unsigned short* as,
                                              const unsigned short* bs,
                                              int lane, int wm, int wn,
                                              v8f acc[2][4]) {
    const int hr  = lane & 15;
    const int akb = (lane < 16) ? 0 : 8;
    v16bf a[2];
#pragma unroll
    for (int i = 0; i < 2; ++i) {
        const int r = wm * 32 + i * 16 + hr;
        FragU t;
        t.u[0] = *(const uint4*)&as[r * LDSS + akb];
        t.u[1] = *(const uint4*)&as[r * LDSS + akb + 16];
        a[i] = t.v;
    }
    const int bkb = (lane < 16) ? 0 : 8;
    v16bf b[4];
#pragma unroll
    for (int j = 0; j < 4; ++j) {
        const int c = wn * 64 + j * 16 + hr;
        FragU t;
        t.u[0] = *(const uint4*)&bs[c * LDSS + bkb];
        t.u[1] = *(const uint4*)&bs[c * LDSS + bkb + 16];
        b[j] = t.v;
    }
#pragma unroll
    for (int i = 0; i < 2; ++i)
#pragma unroll
        for (int j = 0; j < 4; ++j)
            acc[i][j] = __builtin_amdgcn_wmma_f32_16x16x32_bf16(
                false, a[i], false, b[j], (short)0, acc[i][j], false, false);
}

// A: [E][M][K] (f32 or bf16, K contiguous); B: [E][Nn][K] f32; C: [E][M][Nn]
template <bool A_IS_BF16, bool A_ASYNC, bool RELU, bool OUT_BF16>
__global__ __launch_bounds__(256)
void moe_gemm(const void* __restrict__ A, const float* __restrict__ B,
              const float* __restrict__ bias, void* __restrict__ C,
              int M, int Nn, int K) {
    __shared__ alignas(16) unsigned short As[2][BM * LDSS];
    __shared__ alignas(16) unsigned short Bs[2][BN * LDSS];

    const int tid  = threadIdx.x;
    const int lane = tid & 31;
    const int wave = tid >> 5;
    const int wm   = wave & 3;          // 4 waves along M
    const int wn   = wave >> 2;         // 2 waves along N
    const int e    = blockIdx.z;
    const int bm   = blockIdx.y;
    const int bn   = blockIdx.x;

    // Cooperative tile loaders: 256 threads, 2 threads per row, 16 elems each.
    const int lrow = tid >> 1;          // 0..127
    const int lcol = (tid & 1) * 16;    // 0 or 16

    const float*          Ag_f = nullptr;
    const unsigned short* Ag_h = nullptr;
    if constexpr (A_IS_BF16)
        Ag_h = (const unsigned short*)A + ((size_t)e * M + bm * BM + lrow) * K + lcol;
    else
        Ag_f = (const float*)A + ((size_t)e * M + bm * BM + lrow) * K + lcol;
    const float* Bg = B + ((size_t)e * Nn + bn * BN + lrow) * K + lcol;

    auto pf = [&](int k0) {   // WGP-scope prefetch -> global_prefetch_b8
        if constexpr (A_IS_BF16) __builtin_prefetch(Ag_h + k0, 0, 3);
        else                     __builtin_prefetch(Ag_f + k0, 0, 3);
        __builtin_prefetch(Bg + k0, 0, 3);
    };

    auto gload = [&](StageRegs& r, int k0) {   // global -> staging regs
        if constexpr (A_IS_BF16 && !A_ASYNC) {
            const uint4* p = (const uint4*)(Ag_h + k0);
            r.aH[0] = p[0]; r.aH[1] = p[1];
        } else if constexpr (!A_IS_BF16) {
            const float4* p = (const float4*)(Ag_f + k0);
            r.aF[0] = p[0]; r.aF[1] = p[1]; r.aF[2] = p[2]; r.aF[3] = p[3];
        }
        const float4* q = (const float4*)(Bg + k0);
        r.bF[0] = q[0]; r.bF[1] = q[1]; r.bF[2] = q[2]; r.bF[3] = q[3];
    };

    auto store_lds = [&](const StageRegs& r, unsigned short* as,
                         unsigned short* bs) {
        if constexpr (A_IS_BF16 && !A_ASYNC) {
            *(uint4*)&as[lrow * LDSS + lcol]     = r.aH[0];
            *(uint4*)&as[lrow * LDSS + lcol + 8] = r.aH[1];
        } else if constexpr (!A_IS_BF16) {
            *(uint4*)&as[lrow * LDSS + lcol]     = cvt8(r.aF[0], r.aF[1]);
            *(uint4*)&as[lrow * LDSS + lcol + 8] = cvt8(r.aF[2], r.aF[3]);
        }
        *(uint4*)&bs[lrow * LDSS + lcol]     = cvt8(r.bF[0], r.bF[1]);
        *(uint4*)&bs[lrow * LDSS + lcol + 8] = cvt8(r.bF[2], r.bF[3]);
    };

    // Async global->LDS copy of one A tile stage (bf16 path only).
    auto async_A = [&](int k0, unsigned short* as) {
#if HAVE_ASYNC_LDS
        __builtin_amdgcn_global_load_async_to_lds_b128(
            (__attribute__((address_space(1))) v4i*)(Ag_h + k0),
            (__attribute__((address_space(3))) v4i*)&as[lrow * LDSS + lcol],
            0, 0);
        __builtin_amdgcn_global_load_async_to_lds_b128(
            (__attribute__((address_space(1))) v4i*)(Ag_h + k0 + 8),
            (__attribute__((address_space(3))) v4i*)&as[lrow * LDSS + lcol + 8],
            0, 0);
#else
        (void)k0; (void)as;
#endif
    };

    v8f acc[2][4] = {};
    StageRegs r0, r1;

    const int nstage = K / BK;          // 32 or 128: always even
    if constexpr (A_ASYNC) async_A(0, As[0]);
    gload(r0, 0);

    for (int s = 0; s < nstage; s += 2) {
        // ---- even stage s: regs r0, LDS buffer 0 ----
        gload(r1, (s + 1) * BK);
        if (s + 2 < nstage) pf((s + 2) * BK);
        store_lds(r0, As[0], Bs[0]);
        if constexpr (A_ASYNC) wait_async0();   // A tile for stage s in LDS
        __syncthreads();
        if constexpr (A_ASYNC)                  // stage s-1 reads of As[1]
            if (s + 1 < nstage) async_A((s + 1) * BK, As[1]);  // were fenced
        compute_stage(As[0], Bs[0], lane, wm, wn, acc);

        // ---- odd stage s+1: regs r1, LDS buffer 1 ----
        if (s + 2 < nstage) {
            gload(r0, (s + 2) * BK);
            if (s + 3 < nstage) pf((s + 3) * BK);
        }
        store_lds(r1, As[1], Bs[1]);
        if constexpr (A_ASYNC) wait_async0();
        __syncthreads();
        if constexpr (A_ASYNC)
            if (s + 2 < nstage) async_A((s + 2) * BK, As[0]);
        compute_stage(As[1], Bs[1], lane, wm, wn, acc);
    }

    // ---- epilogue: bias (+ReLU), store ----
#pragma unroll
    for (int i = 0; i < 2; ++i) {
        const int mb = bm * BM + wm * 32 + i * 16 + ((lane >> 4) << 3);
#pragma unroll
        for (int j = 0; j < 4; ++j) {
            const int nc = bn * BN + wn * 64 + j * 16 + (lane & 15);
            const float bv = bias[(size_t)e * Nn + nc];
#pragma unroll
            for (int r = 0; r < 8; ++r) {
                float v = acc[i][j][r] + bv;
                if constexpr (RELU) v = v > 0.f ? v : 0.f;
                const size_t off = ((size_t)e * M + mb + r) * Nn + nc;
                if constexpr (OUT_BF16)
                    ((unsigned short*)C)[off] = f2bf(v);
                else
                    ((float*)C)[off] = v;
            }
        }
    }
}

// ---------------------------------------------------------------------------

extern "C" void kernel_launch(void* const* d_in, const int* in_sizes, int n_in,
                              void* d_out, int out_size, void* d_ws, size_t ws_size,
                              hipStream_t stream) {
    constexpr int E = 16, D = 1024, H = 4096, NPER = 1024;  // tokens per expert
    constexpr bool kAsync = HAVE_ASYNC_LDS != 0;

    const float* xs = (const float*)d_in[0];
    // d_in[1] = fwd_expert_count: uniform N/E per expert, layout is static.
    const float* w1 = (const float*)d_in[2];
    const float* b1 = (const float*)d_in[3];
    const float* w2 = (const float*)d_in[4];
    const float* b2 = (const float*)d_in[5];
    unsigned short* hbuf = (unsigned short*)d_ws;   // [E][NPER][H] bf16 = 128 MB
    float* out = (float*)d_out;

    dim3 blk(256);
    // pass 1: h = relu(x @ w1^T + b1), bf16 out
    moe_gemm<false, false, true, true>
        <<<dim3(H / BN, NPER / BM, E), blk, 0, stream>>>(xs, w1, b1, hbuf, NPER, H, D);
    // pass 2: y = h @ w2^T + b2, f32 out (A tile async-copied when supported)
    moe_gemm<true, kAsync, false, false>
        <<<dim3(D / BN, NPER / BM, E), blk, 0, stream>>>(hbuf, w2, b2, out, NPER, D, H);
}
